// unit_gcn_6330781795030
// MI455X (gfx1250) — compile-verified, matrix operand order checked by reading, not run on hardware
//
#include <hip/hip_runtime.h>

typedef __attribute__((ext_vector_type(16))) __bf16 v16bf;
typedef __attribute__((ext_vector_type(8)))  __bf16 v8bf;
typedef __attribute__((ext_vector_type(8)))  float  v8f;

#define NN 64
#define CC 64
#define TT 300
#define VV 25
#define OO 64
#define NA 3
#define TB 16                       // t-values per workgroup
#define VP 32                       // V padded to 32 (one bf16 WMMA K-step)
#define NTB ((TT + TB - 1) / TB)    // 19 t-blocks
#define XSTR (CC + 8)               // xs_t row stride: 72 el = 144 B (bank-friendly)
#define USTR (TB * VP + 8)          // ut row stride: 520 el = 1040 B
#define ASTR (VP + 8)               // AsT row stride: 40 el = 80 B
#define BN_EPS 1e-5f

// Load one 16x32 bf16 fragment where each lane's K-elements are contiguous:
// lane L -> row (L&15), two 16-byte runs at k = h*8 and k = 16 + h*8 (h = L>>4).
// Serves A-fragments (row = M) and B-fragments read from an N-major (transposed)
// buffer (row = N). Emits 2x ds_load_b128.
__device__ __forceinline__ v16bf ld_frag(const __bf16* base, int stride_el) {
  const int L = threadIdx.x & 31;
  const int row = L & 15;
  const int h = (L >> 4) & 1;
  const __bf16* p0 = base + row * stride_el + h * 8;
  v8bf lo = *(const v8bf*)(p0);
  v8bf hi = *(const v8bf*)(p0 + 16);
  return __builtin_shufflevector(lo, hi, 0, 1, 2, 3, 4, 5, 6, 7,
                                 8, 9, 10, 11, 12, 13, 14, 15);
}

// B-fragment of W^T (K = c, N = o) straight from global f32 W[a][o][c]
// via aligned float4 loads (W is tiny and L2-resident), converted to bf16.
__device__ __forceinline__ v16bf ld_wfrag(const float* Wa_o0, int kstep) {
  const int L = threadIdx.x & 31;
  const int n = L & 15;
  const int h = (L >> 4) & 1;
  const float* p0 = Wa_o0 + n * CC + kstep * 32 + h * 8;
  const float4 f0 = *(const float4*)(p0);
  const float4 f1 = *(const float4*)(p0 + 4);
  const float4 f2 = *(const float4*)(p0 + 16);
  const float4 f3 = *(const float4*)(p0 + 20);
  v16bf f;
  f[0]  = (__bf16)f0.x; f[1]  = (__bf16)f0.y; f[2]  = (__bf16)f0.z; f[3]  = (__bf16)f0.w;
  f[4]  = (__bf16)f1.x; f[5]  = (__bf16)f1.y; f[6]  = (__bf16)f1.z; f[7]  = (__bf16)f1.w;
  f[8]  = (__bf16)f2.x; f[9]  = (__bf16)f2.y; f[10] = (__bf16)f2.z; f[11] = (__bf16)f2.w;
  f[12] = (__bf16)f3.x; f[13] = (__bf16)f3.y; f[14] = (__bf16)f3.z; f[15] = (__bf16)f3.w;
  return f;
}

// ---------------- fused GCN kernel ----------------
// Per workgroup: one (n, t-block).  y[n,o,t,w] = sum_a W_a @ (x_n @ A_a).
// Stage 1 (flipped): u^T[v,o] = x^T[v,c] @ W^T[c,o]   (K = C = 64)
// Stage 2 (flipped): y^T[w,o] += A^T[w,v] @ u^T[v,o]  (K = VP = 32)
// Bias b dropped: a per-channel constant cancels exactly under BatchNorm.
__global__ __launch_bounds__(256, 1)
void gcn_fused(const float* __restrict__ x, const float* __restrict__ A,
               const float* __restrict__ W, float* __restrict__ y,
               float* __restrict__ stats) {
  __shared__ __align__(16) __bf16 xs_t[TB * VP][XSTR];  // 73.7 KB  x^T tile
  __shared__ __align__(16) __bf16 ut[OO][USTR];         // 66.6 KB  u  (o rows, v contiguous)
  __shared__ __align__(16) __bf16 AsT[NA][VP][ASTR];    // 7.7 KB   A^T, zero-padded
  __shared__ float lsum[OO];
  __shared__ float lsq[OO];

  const int tid  = threadIdx.x;
  const int wave = tid >> 5;
  const int lane = tid & 31;
  const int ot   = wave >> 1;     // o-tile: columns ot*16 .. ot*16+15
  const int p    = wave & 1;      // row parity: v-tile (stage1) / w-tile (stage2)
  const int nl   = lane & 15;
  const int h    = (lane >> 4) & 1;

  const int n      = blockIdx.x / NTB;
  const int tb     = blockIdx.x % NTB;
  const int t0     = tb * TB;
  const int tcount = min(TB, TT - t0);

  if (tid < OO) { lsum[tid] = 0.f; lsq[tid] = 0.f; }

  // ---- zero xs_t (padded rows / tail-t must be exactly 0, not garbage) ----
  {
    __bf16* flat = &xs_t[0][0];
    const int nel = TB * VP * XSTR;
    v8bf z = {};
    for (int i = tid * 8; i < nel; i += 256 * 8) *(v8bf*)(flat + i) = z;
  }
  // ---- A^T, zero-padded 25 -> 32 (padded K contributes 0 in stage 2) ----
  for (int i = tid; i < NA * VP * VP; i += 256) {
    const int a = i / (VP * VP);
    const int rest = i % (VP * VP);
    const int w = rest / VP;
    const int v = rest % VP;
    float val = 0.f;
    if (v < VV && w < VV) val = A[(a * VV + v) * VV + w];
    AsT[a][w][v] = (__bf16)val;
  }
  __syncthreads();

  // ---- stage x^T: lanes sweep (t,v) columns (coalesced b32 global reads),
  //      each item packs 8 channels -> one ds_store_b128.
  //      All divisors are compile-time constants (no runtime div sequences). ----
  const float* xb = x + (size_t)n * CC * (TT * VV);
  for (int i = tid; i < (TB * VV) * (CC / 8); i += 256) {
    const int col = i % (TB * VV);
    const int cg  = i / (TB * VV);
    const int t   = col / VV;
    const int v   = col % VV;
    if (t < tcount) {
      const float* src = xb + (size_t)(cg * 8) * (TT * VV) + (t0 + t) * VV + v;
      v8bf pk;
#pragma unroll
      for (int j = 0; j < 8; ++j) pk[j] = (__bf16)src[(size_t)j * (TT * VV)];
      *(v8bf*)(&xs_t[t * VP + v][cg * 8]) = pk;
    }
  }
  __syncthreads();

  v8f acc[TB];
#pragma unroll
  for (int t = 0; t < TB; ++t) {
#pragma unroll
    for (int k = 0; k < 8; ++k) acc[t][k] = 0.f;
  }

  for (int a = 0; a < NA; ++a) {
    const float* Wa = W + (size_t)a * OO * CC + (size_t)(ot * 16) * CC;
    const v16bf wb0 = ld_wfrag(Wa, 0);
    const v16bf wb1 = ld_wfrag(Wa, 1);
    __syncthreads();   // previous branch done reading ut

    // ---- stage 1: u^T tiles [v rows p*16..][o cols ot*16..], all t.
    //      Both K-step fragments are issued before the first WMMA so the
    //      second WMMA's operands load under the first WMMA's execution. ----
#pragma unroll
    for (int t = 0; t < TB; ++t) {
      const v16bf xa0 = ld_frag(&xs_t[t * VP + p * 16][0], XSTR);
      const v16bf xa1 = ld_frag(&xs_t[t * VP + p * 16][32], XSTR);
      v8f u = {};
      u = __builtin_amdgcn_wmma_f32_16x16x32_bf16(false, xa0, false, wb0,
                                                  (short)0, u, false, false);
      u = __builtin_amdgcn_wmma_f32_16x16x32_bf16(false, xa1, false, wb1,
                                                  (short)0, u, false, false);
      // D rows are v -> each lane's 8 results are consecutive: 1x ds_store_b128
      v8bf pk;
#pragma unroll
      for (int r = 0; r < 8; ++r) pk[r] = (__bf16)u[r];
      *(v8bf*)(&ut[ot * 16 + nl][t * VP + p * 16 + 8 * h]) = pk;
    }
    __syncthreads();

    // ---- stage 2: y^T tiles [w rows p*16..][o cols ot*16..].
    //      Two independent t's jammed per step to overlap loads with WMMA. ----
    const v16bf aA = ld_frag(&AsT[a][p * 16][0], ASTR);
#pragma unroll
    for (int t = 0; t < TB; t += 2) {
      const v16bf ub0 = ld_frag(&ut[ot * 16][t * VP], USTR);
      const v16bf ub1 = ld_frag(&ut[ot * 16][(t + 1) * VP], USTR);
      acc[t]     = __builtin_amdgcn_wmma_f32_16x16x32_bf16(false, aA, false, ub0,
                                                           (short)0, acc[t], false, false);
      acc[t + 1] = __builtin_amdgcn_wmma_f32_16x16x32_bf16(false, aA, false, ub1,
                                                           (short)0, acc[t + 1], false, false);
    }
  }

  // ---- store raw y + per-channel statistics (one channel per thread) ----
  const int o = ot * 16 + nl;
  float ssum = 0.f, ssq = 0.f;
  float* yb = y + (size_t)n * OO * (TT * VV) + (size_t)o * (TT * VV);
#pragma unroll
  for (int t = 0; t < TB; ++t) {
    if (t < tcount) {
#pragma unroll
      for (int r = 0; r < 8; ++r) {
        const int w = p * 16 + 8 * h + r;   // 8 consecutive w per lane
        if (w < VV) {
          const float val = acc[t][r];
          yb[(size_t)(t0 + t) * VV + w] = val;
          ssum += val;
          ssq  += val * val;
        }
      }
    }
  }
  atomicAdd(&lsum[o], ssum);   // ds_add_f32, 4 threads per channel
  atomicAdd(&lsq[o],  ssq);
  __syncthreads();
  if (tid < OO) {
    atomicAdd(&stats[tid],      lsum[tid]);
    atomicAdd(&stats[OO + tid], lsq[tid]);
  }
}

// ---------------- stats reset (graph-replay safe) ----------------
__global__ void zero_stats(float* stats) {
  if (threadIdx.x < 2 * OO) stats[threadIdx.x] = 0.f;
}

// ---------------- fold sums into BN scale/shift ----------------
__global__ void finalize_stats(float* stats, const float* __restrict__ gamma,
                               const float* __restrict__ beta) {
  const int o = threadIdx.x;
  if (o < OO) {
    const float cnt  = (float)NN * (float)TT * (float)VV;
    const float mean = stats[o] / cnt;
    float var = stats[OO + o] / cnt - mean * mean;
    var = var < 0.f ? 0.f : var;
    const float inv = rsqrtf(var + BN_EPS);
    const float sc  = gamma[o] * inv;
    stats[2 * OO + o] = sc;
    stats[3 * OO + o] = beta[o] - mean * sc;
  }
}

// ---------------- in-place BN + ReLU (float4: plane stride 30000 B is 16B-aligned) --
__global__ __launch_bounds__(256)
void bn_relu(float4* __restrict__ y4, const float* __restrict__ stats) {
  const int    q4     = (TT * VV) / 4;                 // 1875 float4 per (n,o) plane
  const size_t total  = (size_t)NN * OO * q4;
  const size_t stride = (size_t)gridDim.x * blockDim.x;
  for (size_t i = (size_t)blockIdx.x * blockDim.x + threadIdx.x; i < total; i += stride) {
    const int o = (int)((i / (size_t)q4) % OO);
    const float sc = stats[2 * OO + o];
    const float sh = stats[3 * OO + o];
    float4 v = y4[i];
    v.x = fmaxf(fmaf(v.x, sc, sh), 0.f);
    v.y = fmaxf(fmaf(v.y, sc, sh), 0.f);
    v.z = fmaxf(fmaf(v.z, sc, sh), 0.f);
    v.w = fmaxf(fmaf(v.w, sc, sh), 0.f);
    y4[i] = v;
  }
}

extern "C" void kernel_launch(void* const* d_in, const int* in_sizes, int n_in,
                              void* d_out, int out_size, void* d_ws, size_t ws_size,
                              hipStream_t stream) {
  const float* x     = (const float*)d_in[0];
  const float* A     = (const float*)d_in[1];
  const float* W     = (const float*)d_in[2];
  // d_in[3] = b : per-channel constant, cancelled exactly by BatchNorm -> unused
  const float* gamma = (const float*)d_in[4];
  const float* beta  = (const float*)d_in[5];
  float* out   = (float*)d_out;
  float* stats = (float*)d_ws;   // [sum(64) | sumsq(64) | scale(64) | shift(64)]

  zero_stats<<<1, 128, 0, stream>>>(stats);
  gcn_fused<<<NN * NTB, 256, 0, stream>>>(x, A, W, out, stats);
  finalize_stats<<<1, 64, 0, stream>>>(stats, gamma, beta);
  bn_relu<<<4096, 256, 0, stream>>>((float4*)out, stats);
}
